// TransformerBlock_62551903699098
// MI455X (gfx1250) — 16-bit and 8-bit matrix operand order repaired and checked by reading, compiled, not run on hardware
//
#include <hip/hip_runtime.h>
#include <hip/hip_bf16.h>

typedef __attribute__((ext_vector_type(8)))  int       v8i_t;
typedef __attribute__((ext_vector_type(8)))  float     v8f_t;
typedef __attribute__((ext_vector_type(16))) _Float16  v16h_t;

#define DEV_INLINE __device__ __forceinline__

DEV_INLINE void async_copy16(void* lds, const void* g) {
  unsigned           l  = (unsigned)(size_t)lds;
  unsigned long long ga = (unsigned long long)g;
  asm volatile("global_load_async_to_lds_b128 %0, %1, off"
               :: "v"(l), "v"(ga) : "memory");
}
DEV_INLINE void async_wait0() {
  asm volatile("s_wait_asynccnt 0x0" ::: "memory");
}


DEV_INLINE v8i_t load_a_frag_i8(const char* lds, int row0, int stride, int lane) {
  int m    = row0 + (lane & 15);
  int koff = (lane & 16) ? 8 : 0;
  const int* p = (const int*)(lds + m * stride);
  v8i_t f;
#pragma unroll
  for (int q = 0; q < 4; ++q) {
    int kb = q * 16 + koff;
    f[2 * q]     = p[(kb >> 2)];
    f[2 * q + 1] = p[(kb >> 2) + 1];
  }
  return f;
}

DEV_INLINE v8i_t load_b_frag_i8(const char* lds, int col0, int stride, int lane) {
  int n     = col0 + (lane & 15);
  int khalf = (lane & 16) ? 8 : 0;
  const int* p = (const int*)(lds + n * stride);
  v8i_t f;
#pragma unroll
  for (int d = 0; d < 4; ++d) f[d]     = p[(khalf + (d >> 1) * 16 + (d & 1) * 4) >> 2];
#pragma unroll
  for (int d = 0; d < 4; ++d) f[4 + d] = p[(32 + khalf + (d >> 1) * 16 + (d & 1) * 4) >> 2];
  return f;
}

DEV_INLINE v16h_t load_a_frag_f16(const _Float16* p, int kc, int lane) {
  int m    = lane & 15;
  int base = kc * 32 + ((lane & 16) ? 8 : 0);
  const _Float16* row = p + m * 64;
  v16h_t f;
#pragma unroll
  for (int i = 0; i < 8; ++i) f[i]     = row[base + i];
#pragma unroll
  for (int i = 0; i < 8; ++i) f[8 + i] = row[base + 16 + i];
  return f;
}

DEV_INLINE v16h_t load_b_frag_f16(const _Float16* v, int kc, int col0, int lane) {
  int n  = col0 + (lane & 15);
  int kb = kc * 32 + ((lane & 16) ? 8 : 0);
  const _Float16* col = v + n * 64 + kb;
  v16h_t f;
#pragma unroll
  for (int i = 0; i < 16; ++i) f[i] = col[(i < 8) ? i : i + 8];
  return f;
}

DEV_INLINE float qscale(const float* amax) {
  return fmaxf(amax[0] * (1.0f / 127.0f), 1e-8f);
}

__global__ __launch_bounds__(256) void ln_kernel(
    const float* __restrict__ x, const float* __restrict__ g,
    const float* __restrict__ b, float* __restrict__ y, int D) {
  __shared__ float sm[16], sv[16];
  int row = blockIdx.x;
  const float* xr = x + (size_t)row * D;
  float s = 0.f, s2 = 0.f;
  for (int i = threadIdx.x; i < D; i += blockDim.x) {
    float v = xr[i]; s += v; s2 += v * v;
  }
  for (int o = 16; o > 0; o >>= 1) { s += __shfl_xor(s, o); s2 += __shfl_xor(s2, o); }
  int w = threadIdx.x >> 5;
  if ((threadIdx.x & 31) == 0) { sm[w] = s; sv[w] = s2; }
  __syncthreads();
  if (threadIdx.x == 0) {
    float ts = 0.f, t2 = 0.f;
    for (int i = 0; i < (int)(blockDim.x >> 5); ++i) { ts += sm[i]; t2 += sv[i]; }
    float mean = ts / D;
    sm[15] = mean; sv[15] = t2 / D - mean * mean;
  }
  __syncthreads();
  float mean = sm[15];
  float inv  = rsqrtf(sv[15] + 1e-6f);
  for (int i = threadIdx.x; i < D; i += blockDim.x)
    y[(size_t)row * D + i] = (xr[i] - mean) * inv * g[i] + b[i];
}

__global__ __launch_bounds__(256) void add_ln_kernel(
    const float* __restrict__ a, const float* __restrict__ r,
    const float* __restrict__ g, const float* __restrict__ b,
    float* __restrict__ y, int D) {
  __shared__ float sm[16], sv[16];
  int row = blockIdx.x;
  size_t base = (size_t)row * D;
  float s = 0.f, s2 = 0.f;
  for (int i = threadIdx.x; i < D; i += blockDim.x) {
    float t = a[base + i] + r[base + i];
    y[base + i] = t; s += t; s2 += t * t;
  }
  for (int o = 16; o > 0; o >>= 1) { s += __shfl_xor(s, o); s2 += __shfl_xor(s2, o); }
  int w = threadIdx.x >> 5;
  if ((threadIdx.x & 31) == 0) { sm[w] = s; sv[w] = s2; }
  __syncthreads();
  if (threadIdx.x == 0) {
    float ts = 0.f, t2 = 0.f;
    for (int i = 0; i < (int)(blockDim.x >> 5); ++i) { ts += sm[i]; t2 += sv[i]; }
    float mean = ts / D;
    sm[15] = mean; sv[15] = t2 / D - mean * mean;
  }
  __syncthreads();
  float mean = sm[15];
  float inv  = rsqrtf(sv[15] + 1e-6f);
  for (int i = threadIdx.x; i < D; i += blockDim.x)
    y[base + i] = (y[base + i] - mean) * inv * g[i] + b[i];
}

__global__ __launch_bounds__(256) void absmax_kernel(
    const float* __restrict__ x, size_t n, unsigned int* __restrict__ amax) {
  float m = 0.f;
  for (size_t i = (size_t)blockIdx.x * blockDim.x + threadIdx.x; i < n;
       i += (size_t)gridDim.x * blockDim.x)
    m = fmaxf(m, fabsf(x[i]));
  for (int o = 16; o > 0; o >>= 1) m = fmaxf(m, __shfl_xor(m, o));
  if ((threadIdx.x & 31) == 0) atomicMax(amax, __float_as_uint(m));
}

__global__ __launch_bounds__(256) void quant_kernel(
    const float* __restrict__ x, size_t n, const float* __restrict__ amax,
    signed char* __restrict__ q) {
  float inv = 1.0f / qscale(amax);
  for (size_t i = (size_t)blockIdx.x * blockDim.x + threadIdx.x; i < n;
       i += (size_t)gridDim.x * blockDim.x) {
    float v = fminf(fmaxf(x[i] * inv, -127.f), 127.f);
    q[i] = (signed char)__float2int_rn(v);
  }
}

__global__ __launch_bounds__(256) void quantT_kernel(
    const float* __restrict__ W, const float* __restrict__ amax,
    signed char* __restrict__ out, int K, int N) {
  __shared__ float tile[32][33];
  int kt = blockIdx.y * 32, nt = blockIdx.x * 32;
  float inv = 1.0f / qscale(amax);
  for (int r = threadIdx.y; r < 32; r += 8)
    tile[r][threadIdx.x] = W[(size_t)(kt + r) * N + nt + threadIdx.x];
  __syncthreads();
  for (int r = threadIdx.y; r < 32; r += 8) {
    float v = fminf(fmaxf(tile[threadIdx.x][r] * inv, -127.f), 127.f);
    out[(size_t)(nt + r) * K + kt + threadIdx.x] = (signed char)__float2int_rn(v);
  }
}

__global__ __launch_bounds__(256) void quantT_v_kernel(
    const float* __restrict__ vf, const float* __restrict__ amax,
    signed char* __restrict__ out) {
  const int S = 1024, D = 1024, HD = 32, H = 32;
  __shared__ float tile[32][33];
  int st = blockIdx.x * 32, h = blockIdx.y, b = blockIdx.z;
  float inv = 1.0f / qscale(amax);
  for (int r = threadIdx.y; r < 32; r += 8)
    tile[r][threadIdx.x] = vf[((size_t)(b * S + st + r)) * D + h * HD + threadIdx.x];
  __syncthreads();
  for (int r = threadIdx.y; r < 32; r += 8) {
    float v = fminf(fmaxf(tile[threadIdx.x][r] * inv, -127.f), 127.f);
    out[(((size_t)b * H + h) * HD + r) * S + st + threadIdx.x] =
        (signed char)__float2int_rn(v);
  }
}

__global__ __launch_bounds__(256) void gelu_kernel(float* __restrict__ x, size_t n) {
  for (size_t i = (size_t)blockIdx.x * blockDim.x + threadIdx.x; i < n;
       i += (size_t)gridDim.x * blockDim.x) {
    float v = x[i];
    float c = 0.7978845608028654f * (v + 0.044715f * v * v * v);
    x[i] = 0.5f * v * (1.0f + tanhf(c));
  }
}

__global__ __launch_bounds__(256) void gemm_i8_kernel(
    const signed char* __restrict__ A,
    const signed char* __restrict__ Wt,
    const float* __restrict__ amaxA, const float* __restrict__ amaxW,
    const float* __restrict__ bias, float* __restrict__ C,
    int M, int K, int N) {
  const int STR = 80;
  __shared__ __align__(16) char lds_a[2][128 * 80];
  __shared__ __align__(16) char lds_b[2][128 * 80];
  int tid = threadIdx.x, lane = tid & 31, wave = tid >> 5;
  int m0 = blockIdx.y * 128, n0 = blockIdx.x * 128;
  int wr = (wave & 3) * 32;
  int wc = (wave >> 2) * 64;

  int row0 = tid >> 2, row1 = row0 + 64, seg = (tid & 3) * 16;
  const signed char* pa0 = A  + (size_t)(m0 + row0) * K + seg;
  const signed char* pa1 = A  + (size_t)(m0 + row1) * K + seg;
  const signed char* pb0 = Wt + (size_t)(n0 + row0) * K + seg;
  const signed char* pb1 = Wt + (size_t)(n0 + row1) * K + seg;
  unsigned la0 = row0 * STR + seg, la1 = row1 * STR + seg;

  auto stage = [&](int buf, int kt) {
    async_copy16(&lds_a[buf][la0], pa0 + kt);
    async_copy16(&lds_a[buf][la1], pa1 + kt);
    async_copy16(&lds_b[buf][la0], pb0 + kt);
    async_copy16(&lds_b[buf][la1], pb1 + kt);
  };

  v8i_t acc[2][4];
#pragma unroll
  for (int i = 0; i < 2; ++i)
#pragma unroll
    for (int j = 0; j < 4; ++j) acc[i][j] = {};

  int nT = K >> 6;
  stage(0, 0);
  for (int t = 0; t < nT; ++t) {
    int cur = t & 1;
    async_wait0();
    __syncthreads();
    if (t + 1 < nT) stage(1 - cur, (t + 1) << 6);

    v8i_t af[2], bf[4];
#pragma unroll
    for (int i = 0; i < 2; ++i)
      af[i] = load_a_frag_i8(lds_a[cur], wr + 16 * i, STR, lane);
#pragma unroll
    for (int j = 0; j < 4; ++j)
      bf[j] = load_b_frag_i8(lds_b[cur], wc + 16 * j, STR, lane);
#pragma unroll
    for (int i = 0; i < 2; ++i)
#pragma unroll
      for (int j = 0; j < 4; ++j)
        acc[i][j] = __builtin_amdgcn_wmma_i32_16x16x64_iu8(true, af[i], true, bf[j],
                                                           acc[i][j], false, false);
  }

  float s = qscale(amaxA) * qscale(amaxW);
  int col    = lane & 15;
  int rowadd = (lane & 16) ? 8 : 0;
#pragma unroll
  for (int i = 0; i < 2; ++i)
#pragma unroll
    for (int j = 0; j < 4; ++j)
#pragma unroll
      for (int r = 0; r < 8; ++r) {
        int m  = m0 + wr + 16 * i + r + rowadd;
        int n1 = n0 + wc + 16 * j + col;
        C[(size_t)m * N + n1] = (float)acc[i][j][r] * s + bias[n1];
      }
}

__global__ __launch_bounds__(128) void attn_kernel(
    const signed char* __restrict__ q8,
    const signed char* __restrict__ k8,
    const signed char* __restrict__ v8t,
    const float* __restrict__ amax_q, const float* __restrict__ amax_k,
    const float* __restrict__ amax_v,
    const float* __restrict__ rel,
    float* __restrict__ out) {
  const int S = 1024, D = 1024, HD = 32, H = 32, P = 32;
  int b = blockIdx.z, h = blockIdx.y, qb = blockIdx.x;
  int tid = threadIdx.x, lane = tid & 31, wave = tid >> 5;

  __shared__ __align__(16) char     lds_q[64 * 64];
  __shared__ __align__(16) char     lds_k[2][64 * 64];
  __shared__ __align__(16) _Float16 lds_v[2][32 * 64];
  __shared__ __align__(16) _Float16 lds_p[4][16 * 64];
  __shared__ float lds_rel[65];

  float sq = qscale(amax_q), sk = qscale(amax_k), sv = qscale(amax_v);
  float qkscale = sq * sk;

  if (tid < 65) lds_rel[tid] = rel[tid * HD + h];

  int krow = tid >> 1, khalf = tid & 1;
  const signed char* pk = k8 + ((size_t)(b * S + krow)) * D + h * HD + khalf * 16;
  unsigned lk = krow * 64 + khalf * 16;
  int vd = tid >> 2, vseg = (tid & 3) * 16;
  const signed char* pv = v8t + (((size_t)b * H + h) * HD + vd) * S + vseg;

  {
    int4 z = {0, 0, 0, 0};
    *(int4*)(lds_q + krow * 64 + 32 + khalf * 16)    = z;
    *(int4*)(lds_k[0] + krow * 64 + 32 + khalf * 16) = z;
    *(int4*)(lds_k[1] + krow * 64 + 32 + khalf * 16) = z;
    async_copy16(lds_q + krow * 64 + khalf * 16,
                 q8 + ((size_t)(b * S + qb * 64 + krow)) * D + h * HD + khalf * 16);
  }

  auto stage_kv = [&](int buf, int kt) {
    async_copy16(&lds_k[buf][lk], pk + (size_t)kt * D);
    int4 vv = *(const int4*)(pv + kt);
    const signed char* vb = (const signed char*)&vv;
    _Float16* dst = &lds_v[buf][vd * 64 + vseg];
#pragma unroll
    for (int j = 0; j < 16; ++j) dst[j] = (_Float16)((float)vb[j] * sv);
  };

  stage_kv(0, 0);
  async_wait0();
  __syncthreads();

  v8i_t qfrag = load_a_frag_i8(lds_q, wave * 16, 64, lane);

  v8f_t acc0 = {}, acc1 = {};
  float rmax[8], rsum[8];
#pragma unroll
  for (int r = 0; r < 8; ++r) { rmax[r] = -3.0e38f; rsum[r] = 0.f; }

  int rowadd    = (lane & 16) ? 8 : 0;
  int col       = lane & 15;
  int qrow_base = qb * 64 + wave * 16;

  const int nT = S / 64;
  for (int t = 0; t < nT; ++t) {
    int cur = t & 1;
    if (t) {
      async_wait0();
      __syncthreads();
    }
    if (t + 1 < nT) stage_kv(1 - cur, (t + 1) * 64);

    float sc[4][8];
#pragma unroll
    for (int u = 0; u < 4; ++u) {
      v8i_t kf = load_b_frag_i8(lds_k[cur], u * 16, 64, lane);
      v8i_t z  = {};
      v8i_t si = __builtin_amdgcn_wmma_i32_16x16x64_iu8(true, qfrag, true, kf, z, false, false);
      int key_g = t * 64 + u * 16 + col;
#pragma unroll
      for (int r = 0; r < 8; ++r) {
        int qi  = qrow_base + r + rowadd;
        int dlt = qi - key_g + P;
        dlt = dlt < 0 ? 0 : (dlt > 2 * P ? 2 * P : dlt);
        sc[u][r] = ((float)si[r] * qkscale + lds_rel[dlt]) * (1.0f / 32.0f);
      }
    }

#pragma unroll
    for (int r = 0; r < 8; ++r) {
      float tm = fmaxf(fmaxf(sc[0][r], sc[1][r]), fmaxf(sc[2][r], sc[3][r]));
      for (int o = 1; o < 16; o <<= 1) tm = fmaxf(tm, __shfl_xor(tm, o));
      float nm   = fmaxf(rmax[r], tm);
      float corr = __expf(rmax[r] - nm);
      rmax[r] = nm;
      float es = 0.f;
#pragma unroll
      for (int u = 0; u < 4; ++u) { sc[u][r] = __expf(sc[u][r] - nm); es += sc[u][r]; }
      for (int o = 1; o < 16; o <<= 1) es += __shfl_xor(es, o);
      rsum[r] = rsum[r] * corr + es;
      acc0[r] *= corr;
      acc1[r] *= corr;
    }

#pragma unroll
    for (int u = 0; u < 4; ++u)
#pragma unroll
      for (int r = 0; r < 8; ++r)
        lds_p[wave][(r + rowadd) * 64 + u * 16 + col] = (_Float16)sc[u][r];
    asm volatile("s_wait_dscnt 0" ::: "memory");

#pragma unroll
    for (int kc = 0; kc < 2; ++kc) {
      v16h_t pa  = load_a_frag_f16(&lds_p[wave][0], kc, lane);
      v16h_t vb0 = load_b_frag_f16(lds_v[cur], kc, 0, lane);
      v16h_t vb1 = load_b_frag_f16(lds_v[cur], kc, 16, lane);
      acc0 = __builtin_amdgcn_wmma_f32_16x16x32_f16(false, pa, false, vb0, (short)0, acc0, false, false);
      acc1 = __builtin_amdgcn_wmma_f32_16x16x32_f16(false, pa, false, vb1, (short)0, acc1, false, false);
    }
    __syncthreads();
  }

#pragma unroll
  for (int r = 0; r < 8; ++r) {
    int qi    = qrow_base + r + rowadd;
    float inv = 1.0f / (rsum[r] + 1e-6f);
    size_t base = ((size_t)(b * S) + qi) * D + h * HD;
    out[base + col]      = acc0[r] * inv;
    out[base + 16 + col] = acc1[r] * inv;
  }
}

extern "C" void kernel_launch(void* const* d_in, const int* in_sizes, int n_in,
                              void* d_out, int out_size, void* d_ws, size_t ws_size,
                              hipStream_t stream) {
  (void)in_sizes; (void)n_in; (void)out_size; (void)ws_size;
  const int B = 4, S = 1024, D = 1024, H = 32, F = 4096;
  const size_t NT = (size_t)B * S;

  const float* inputs = (const float*)d_in[0];
  const float* ln1_s  = (const float*)d_in[1];
  const float* ln1_b  = (const float*)d_in[2];
  const float* Wq = (const float*)d_in[3];  const float* bq = (const float*)d_in[4];
  const float* Wk = (const float*)d_in[5];  const float* bk = (const float*)d_in[6];
  const float* Wv = (const float*)d_in[7];  const float* bv = (const float*)d_in[8];
  const float* rel = (const float*)d_in[9];
  const float* Wo = (const float*)d_in[10]; const float* bo = (const float*)d_in[11];
  const float* ln2_s = (const float*)d_in[12];
  const float* ln2_b = (const float*)d_in[13];
  const float* W1 = (const float*)d_in[14]; const float* b1 = (const float*)d_in[15];
  const float* W2 = (const float*)d_in[16]; const float* b2 = (const float*)d_in[17];
  float* out = (float*)d_out;

  char* w = (char*)d_ws;
  unsigned int* amax = (unsigned int*)w;
  size_t off = 256;
  auto F32 = [&](size_t n) { float* p = (float*)(w + off); off += n * 4; return p; };
  auto I8  = [&](size_t n) { signed char* p = (signed char*)(w + off); off += (n + 255) & ~(size_t)255; return p; };

  float*       xln   = F32(NT * D);
  signed char* x8    = I8(NT * D);
  signed char* wq8   = I8((size_t)D * D);
  signed char* wk8   = I8((size_t)D * D);
  signed char* wv8   = I8((size_t)D * D);
  signed char* wo8   = I8((size_t)D * D);
  signed char* w18   = I8((size_t)D * F);
  signed char* w28   = I8((size_t)F * D);
  float*       qf    = F32(NT * D);
  float*       kf    = F32(NT * D);
  float*       vf    = F32(NT * D);
  signed char* q8    = I8(NT * D);
  signed char* k8    = I8(NT * D);
  signed char* v8t   = I8(NT * D);
  float*       attnf = F32(NT * D);
  signed char* attn8 = I8(NT * D);
  float*       attno = F32(NT * D);
  float*       y     = F32(NT * D);
  signed char* y8    = I8(NT * D);
  float*       hbuf  = F32(NT * F);
  signed char* h8    = I8(NT * F);

  const float* amf = (const float*)amax;
  hipMemsetAsync(d_ws, 0, 256, stream);

  ln_kernel<<<(int)NT, 256, 0, stream>>>(inputs, ln1_s, ln1_b, xln, D);

  absmax_kernel<<<1024, 256, 0, stream>>>(xln, NT * D, amax + 0);
  quant_kernel<<<2048, 256, 0, stream>>>(xln, NT * D, amf + 0, x8);
  dim3 tb(32, 8);
  absmax_kernel<<<512, 256, 0, stream>>>(Wq, (size_t)D * D, amax + 1);
  quantT_kernel<<<dim3(D / 32, D / 32), tb, 0, stream>>>(Wq, amf + 1, wq8, D, D);
  absmax_kernel<<<512, 256, 0, stream>>>(Wk, (size_t)D * D, amax + 2);
  quantT_kernel<<<dim3(D / 32, D / 32), tb, 0, stream>>>(Wk, amf + 2, wk8, D, D);
  absmax_kernel<<<512, 256, 0, stream>>>(Wv, (size_t)D * D, amax + 3);
  quantT_kernel<<<dim3(D / 32, D / 32), tb, 0, stream>>>(Wv, amf + 3, wv8, D, D);
  absmax_kernel<<<512, 256, 0, stream>>>(Wo, (size_t)D * D, amax + 4);
  quantT_kernel<<<dim3(D / 32, D / 32), tb, 0, stream>>>(Wo, amf + 4, wo8, D, D);
  absmax_kernel<<<1024, 256, 0, stream>>>(W1, (size_t)D * F, amax + 5);
  quantT_kernel<<<dim3(F / 32, D / 32), tb, 0, stream>>>(W1, amf + 5, w18, D, F);
  absmax_kernel<<<1024, 256, 0, stream>>>(W2, (size_t)F * D, amax + 6);
  quantT_kernel<<<dim3(D / 32, F / 32), tb, 0, stream>>>(W2, amf + 6, w28, F, D);

  dim3 gQKV(D / 128, (int)NT / 128);
  gemm_i8_kernel<<<gQKV, 256, 0, stream>>>(x8, wq8, amf + 0, amf + 1, bq, qf, (int)NT, D, D);
  gemm_i8_kernel<<<gQKV, 256, 0, stream>>>(x8, wk8, amf + 0, amf + 2, bk, kf, (int)NT, D, D);
  gemm_i8_kernel<<<gQKV, 256, 0, stream>>>(x8, wv8, amf + 0, amf + 3, bv, vf, (int)NT, D, D);

  absmax_kernel<<<1024, 256, 0, stream>>>(qf, NT * D, amax + 7);
  quant_kernel<<<2048, 256, 0, stream>>>(qf, NT * D, amf + 7, q8);
  absmax_kernel<<<1024, 256, 0, stream>>>(kf, NT * D, amax + 8);
  quant_kernel<<<2048, 256, 0, stream>>>(kf, NT * D, amf + 8, k8);
  absmax_kernel<<<1024, 256, 0, stream>>>(vf, NT * D, amax + 9);
  quantT_v_kernel<<<dim3(S / 32, H, B), tb, 0, stream>>>(vf, amf + 9, v8t);

  attn_kernel<<<dim3(S / 64, H, B), 128, 0, stream>>>(
      q8, k8, v8t, amf + 7, amf + 8, amf + 9, rel, attnf);

  absmax_kernel<<<1024, 256, 0, stream>>>(attnf, NT * D, amax + 10);
  quant_kernel<<<2048, 256, 0, stream>>>(attnf, NT * D, amf + 10, attn8);
  gemm_i8_kernel<<<gQKV, 256, 0, stream>>>(attn8, wo8, amf + 10, amf + 4, bo, attno, (int)NT, D, D);

  add_ln_kernel<<<(int)NT, 256, 0, stream>>>(attno, inputs, ln2_s, ln2_b, y, D);

  absmax_kernel<<<1024, 256, 0, stream>>>(y, NT * D, amax + 11);
  quant_kernel<<<2048, 256, 0, stream>>>(y, NT * D, amf + 11, y8);
  gemm_i8_kernel<<<dim3(F / 128, (int)NT / 128), 256, 0, stream>>>(
      y8, w18, amf + 11, amf + 5, b1, hbuf, (int)NT, D, F);
  gelu_kernel<<<4096, 256, 0, stream>>>(hbuf, NT * F);
  absmax_kernel<<<2048, 256, 0, stream>>>(hbuf, NT * F, amax + 12);
  quant_kernel<<<4096, 256, 0, stream>>>(hbuf, NT * F, amf + 12, h8);
  gemm_i8_kernel<<<dim3(D / 128, (int)NT / 128), 256, 0, stream>>>(
      h8, w28, amf + 12, amf + 6, b2, out, (int)NT, F, D);
}
